// CurveQueryDecoder_60361470378235
// MI455X (gfx1250) — matrix operand order repaired and checked by reading, not compiled, not run on hardware
//
#include <hip/hip_runtime.h>
#include <hip/hip_bf16.h>
#include <math.h>

typedef __attribute__((ext_vector_type(16))) __bf16 bf16x16;
typedef __attribute__((ext_vector_type(8)))  float  f32x8;

__device__ __forceinline__ float gelu_f(float x) {
    return 0.5f * x * (1.0f + erff(x * 0.70710678118654752f));
}
__device__ __forceinline__ float sigmoid_f(float x) {
    return 1.0f / (1.0f + expf(-x));
}
__device__ __forceinline__ float wave_sum(float v) {
    #pragma unroll
    for (int m = 16; m >= 1; m >>= 1) v += __shfl_xor(v, m, 32);
    return v;
}

// A fragment (16x32 bf16): lanes 0-15 hold K k0+0..7 & k0+16..23; lanes 16-31 the +8 halves
__device__ __forceinline__ bf16x16 load_a_frag(const float* __restrict__ Arow, int k0, int half) {
    float4 a0 = *(const float4*)(Arow + k0 + half * 8);
    float4 a1 = *(const float4*)(Arow + k0 + half * 8 + 4);
    float4 a2 = *(const float4*)(Arow + k0 + 16 + half * 8);
    float4 a3 = *(const float4*)(Arow + k0 + 16 + half * 8 + 4);
    bf16x16 f;
    f[0]=(__bf16)a0.x; f[1]=(__bf16)a0.y; f[2]=(__bf16)a0.z; f[3]=(__bf16)a0.w;
    f[4]=(__bf16)a1.x; f[5]=(__bf16)a1.y; f[6]=(__bf16)a1.z; f[7]=(__bf16)a1.w;
    f[8]=(__bf16)a2.x; f[9]=(__bf16)a2.y; f[10]=(__bf16)a2.z; f[11]=(__bf16)a2.w;
    f[12]=(__bf16)a3.x; f[13]=(__bf16)a3.y; f[14]=(__bf16)a3.z; f[15]=(__bf16)a3.w;
    return f;
}

// B fragment (32x16 bf16): lanes 0-15 hold K k0+0..7 & k0+16..23; lanes 16-31 the +8 halves
__device__ __forceinline__ bf16x16 load_b_frag(const float* __restrict__ Brow, int k0, int half) {
    float4 b0 = *(const float4*)(Brow + k0 + half * 8);
    float4 b1 = *(const float4*)(Brow + k0 + half * 8 + 4);
    float4 b2 = *(const float4*)(Brow + k0 + 16 + half * 8);
    float4 b3 = *(const float4*)(Brow + k0 + 16 + half * 8 + 4);
    bf16x16 f;
    f[0]=(__bf16)b0.x; f[1]=(__bf16)b0.y; f[2]=(__bf16)b0.z; f[3]=(__bf16)b0.w;
    f[4]=(__bf16)b1.x; f[5]=(__bf16)b1.y; f[6]=(__bf16)b1.z; f[7]=(__bf16)b1.w;
    f[8]=(__bf16)b2.x; f[9]=(__bf16)b2.y; f[10]=(__bf16)b2.z; f[11]=(__bf16)b2.w;
    f[12]=(__bf16)b3.x; f[13]=(__bf16)b3.y; f[14]=(__bf16)b3.z; f[15]=(__bf16)b3.w;
    return f;
}

// ---------------------------------------------------------------------------
// Generic bf16-WMMA GEMM:  C[b,m,n] = act( sum_k A[b,m,k]*Bm[b,n,k] + bias[n] )
// One wave computes a 32x64 tile: 2 M-tiles x 4 N-tiles = 8 WMMAs per K-step
// of 32 (each B fragment reused twice). Out-of-range rows/cols are pointer-
// clamped; garbage lanes are never stored, so all loads are unconditional.
// K must be a multiple of 32; rows must be 16B aligned. B is row-major (N x K).
// ---------------------------------------------------------------------------
__global__ __launch_bounds__(32)
void gemm_wmma_kernel(const float* __restrict__ A, long bsA, long lda,
                      const float* __restrict__ Bm, long bsB, long ldbN,
                      float* __restrict__ C, long bsC, long ldc,
                      const float* __restrict__ bias,
                      int M, int N, int K, int act)
{
    const int lane = threadIdx.x;
    const int half = lane >> 4;
    const int lo   = lane & 15;
    const int n0 = blockIdx.x * 64;
    const int m0 = blockIdx.y * 32;
    const int b  = blockIdx.z;
    const float* Ab = A  + (long)b * bsA;
    const float* Bb = Bm + (long)b * bsB;
    float*       Cb = C  + (long)b * bsC;

    int mr0 = m0 + lo;      if (mr0 >= M) mr0 = M - 1;   // clamp: safe, never stored
    int mr1 = m0 + 16 + lo; if (mr1 >= M) mr1 = M - 1;
    const float* Arow0 = Ab + (long)mr0 * lda;
    const float* Arow1 = Ab + (long)mr1 * lda;

    const float* Brow[4];
    #pragma unroll
    for (int t = 0; t < 4; ++t) {
        int nc = n0 + t * 16 + lo; if (nc >= N) nc = N - 1;
        Brow[t] = Bb + (long)nc * ldbN;
    }

    const f32x8 z = {0.f,0.f,0.f,0.f,0.f,0.f,0.f,0.f};
    f32x8 acc0[4], acc1[4];
    #pragma unroll
    for (int t = 0; t < 4; ++t) { acc0[t] = z; acc1[t] = z; }

    for (int k0 = 0; k0 < K; k0 += 32) {
        bf16x16 af0 = load_a_frag(Arow0, k0, half);
        bf16x16 af1 = load_a_frag(Arow1, k0, half);
        #pragma unroll
        for (int t = 0; t < 4; ++t) {
            bf16x16 bv = load_b_frag(Brow[t], k0, half);
            acc0[t] = __builtin_amdgcn_wmma_f32_16x16x32_bf16(false, af0, false, bv,
                                                              (short)0, acc0[t], false, false);
            acc1[t] = __builtin_amdgcn_wmma_f32_16x16x32_bf16(false, af1, false, bv,
                                                              (short)0, acc1[t], false, false);
        }
    }

    #pragma unroll
    for (int t = 0; t < 4; ++t) {
        int c = n0 + t * 16 + lo;
        #pragma unroll
        for (int j = 0; j < 8; ++j) {
            int r0 = m0 + j + half * 8;
            int r1 = r0 + 16;
            if (r0 < M && c < N) {
                float v = acc0[t][j];
                if (bias) v += bias[c];
                if (act == 1)      v = gelu_f(v);
                else if (act == 2) v = sigmoid_f(v);
                Cb[(long)r0 * ldc + c] = v;
            }
            if (r1 < M && c < N) {
                float v = acc1[t][j];
                if (bias) v += bias[c];
                if (act == 1)      v = gelu_f(v);
                else if (act == 2) v = sigmoid_f(v);
                Cb[(long)r1 * ldc + c] = v;
            }
        }
    }
}

// ---------------------------------------------------------------------------
// Tiled LDS transpose: per batch, in (R x Ccols) -> out (Ccols x R)
// ---------------------------------------------------------------------------
__global__ __launch_bounds__(256)
void transpose_kernel(const float* __restrict__ in, float* __restrict__ out,
                      int R, int Ccols) {
    __shared__ float tile[32][33];
    int b = blockIdx.z;
    const float* ib = in + (long)b * R * Ccols;
    float* ob = out + (long)b * R * Ccols;
    int c0 = blockIdx.x * 32;
    int r0 = blockIdx.y * 32;
    int tx = threadIdx.x & 31, ty = threadIdx.x >> 5;
    #pragma unroll
    for (int k = 0; k < 4; ++k)
        tile[ty + 8 * k][tx] = ib[(long)(r0 + ty + 8 * k) * Ccols + c0 + tx];
    __syncthreads();
    #pragma unroll
    for (int k = 0; k < 4; ++k)
        ob[(long)(c0 + ty + 8 * k) * R + r0 + tx] = tile[tx][ty + 8 * k];
}

// ---------------------------------------------------------------------------
// Elementwise / small kernels
// ---------------------------------------------------------------------------
__global__ void bcast_kernel(const float* __restrict__ src, float* __restrict__ dst,
                             int total, int period) {
    int i = blockIdx.x * 256 + threadIdx.x;
    if (i < total) dst[i] = src[i % period];
}

__global__ void add_bcast_kernel(const float* __restrict__ a, const float* __restrict__ p,
                                 float* __restrict__ o, int total, int period) {
    int i = blockIdx.x * 256 + threadIdx.x;
    if (i < total) o[i] = a[i] + p[i % period];
}

__global__ void concat_kernel(const float* __restrict__ a, const float* __restrict__ b,
                              float* __restrict__ o, int total) {
    int i = blockIdx.x * 256 + threadIdx.x;
    if (i >= total) return;
    int r = i >> 9, d = i & 511;
    o[i] = (d < 256) ? a[r * 256 + d] : b[r * 256 + d - 256];
}

__global__ void axpyg_kernel(float* __restrict__ q, const float* __restrict__ g,
                             const float* __restrict__ a, int total) {
    int i = blockIdx.x * 256 + threadIdx.x;
    if (i < total) q[i] += g[i] * a[i];
}

// u[k] = sum_o (sum_d mean(query_feat)[d]*bqW[o,d]) * bmW[o,k]     (one block, 256 thr)
__global__ __launch_bounds__(256)
void bato_u_kernel(const float* __restrict__ qf, const float* __restrict__ bqW,
                   const float* __restrict__ bmW, float* __restrict__ u) {
    __shared__ float qm[256];
    __shared__ float qp[256];
    int t = threadIdx.x;
    float s = 0.f;
    for (int i = 0; i < 300; ++i) s += qf[i * 256 + t];
    qm[t] = s * (1.0f / 300.0f);
    __syncthreads();
    float s2 = 0.f;
    for (int d = 0; d < 256; ++d) s2 += qm[d] * bqW[t * 256 + d];
    qp[t] = s2;
    __syncthreads();
    float s3 = 0.f;
    for (int o = 0; o < 256; ++o) s3 += qp[o] * bmW[o * 256 + t];
    u[t] = s3;
}

// mem = memory * (1 + sigmoid(aff) * gate);  one wave per row
__global__ __launch_bounds__(256)
void bato_apply_kernel(const float* __restrict__ memory, const float* __restrict__ H,
                       const float* __restrict__ g2W, const float* __restrict__ g2b,
                       const float* __restrict__ u, float* __restrict__ mem, long rows) {
    long row = (long)blockIdx.x * 8 + (threadIdx.x >> 5);
    int lane = threadIdx.x & 31;
    if (row >= rows) return;
    const float* mrow = memory + row * 256;
    float aff = 0.f;
    #pragma unroll
    for (int i = 0; i < 8; ++i) { int d = lane + 32 * i; aff += mrow[d] * u[d]; }
    const float* hrow = H + row * 64;
    float g = hrow[lane] * g2W[lane] + hrow[lane + 32] * g2W[lane + 32];
    aff = wave_sum(aff);
    g   = wave_sum(g);
    float gate  = sigmoid_f(g + g2b[0]);
    float sa    = sigmoid_f(aff);
    float scale = 1.0f + sa * gate;
    float* orow = mem + row * 256;
    #pragma unroll
    for (int i = 0; i < 8; ++i) { int d = lane + 32 * i; orow[d] = mrow[d] * scale; }
}

__global__ void rp2_kernel(const float* __restrict__ rp1, const float* __restrict__ W,
                           const float* __restrict__ b2, float* __restrict__ rp) {
    int i = blockIdx.x * 256 + threadIdx.x;
    if (i >= 600) return;
    int q = i >> 1, o = i & 1;
    float s = b2[o];
    for (int j = 0; j < 128; ++j) s += rp1[q * 128 + j] * W[o * 128 + j];
    rp[i] = sigmoid_f(s);
}

__global__ __launch_bounds__(128)
void relbias_kernel(const float* __restrict__ rp, const float* __restrict__ W1,
                    const float* __restrict__ b1, const float* __restrict__ W2,
                    const float* __restrict__ b2, float* __restrict__ bias) {
    int i = blockIdx.x * 128 + threadIdx.x;
    if (i >= 300 * 300) return;
    int q1 = i / 300, q2 = i % 300;
    float dx = rp[q1 * 2 + 0] - rp[q2 * 2 + 0];
    float dy = rp[q1 * 2 + 1] - rp[q2 * 2 + 1];
    float dist = sqrtf(dx * dx + dy * dy + 1e-8f);
    float ang  = atan2f(dy, dx) * 0.3183098861837907f;
    float r2 = logf(dist);
    float h[64];
    #pragma unroll
    for (int j = 0; j < 64; ++j)
        h[j] = gelu_f(b1[j] + W1[j*4+0]*dx + W1[j*4+1]*dy + W1[j*4+2]*r2 + W1[j*4+3]*ang);
    for (int hh = 0; hh < 8; ++hh) {
        float s = b2[hh];
        #pragma unroll
        for (int j = 0; j < 64; ++j) s += W2[hh * 64 + j] * h[j];
        bias[((long)hh * 300 + q1) * 300 + q2] = s;
    }
}

__global__ __launch_bounds__(256)
void l2norm_rows_kernel(float* __restrict__ X, long rows) {
    long row = (long)blockIdx.x * 8 + (threadIdx.x >> 5);
    int lane = threadIdx.x & 31;
    if (row >= rows) return;
    float* p = X + row * 256;
    float v[8]; float ss = 0.f;
    #pragma unroll
    for (int i = 0; i < 8; ++i) { v[i] = p[lane + 32 * i]; ss += v[i] * v[i]; }
    ss = wave_sum(ss);
    float inv = 1.0f / fmaxf(sqrtf(ss), 1e-6f);
    #pragma unroll
    for (int i = 0; i < 8; ++i) p[lane + 32 * i] = v[i] * inv;
}

// top-96 of a 16384 sim row kept LDS-resident + softmax + weighted gather of mem rows
__global__ __launch_bounds__(256)
void topk_align_kernel(const float* __restrict__ sim, const float* __restrict__ mem,
                       float* __restrict__ aligned, int Qn, int Nn, int Kk) {
    extern __shared__ unsigned char smraw[];
    float* sv  = (float*)smraw;          // Nn
    float* rv  = sv + Nn;                // 256
    int*   ri  = (int*)(rv + 256);       // 256
    float* tv  = (float*)(ri + 256);     // Kk
    int*   tix = (int*)(tv + Kk);        // Kk
    float* w   = (float*)(tix + Kk);     // Kk
    int q = blockIdx.x, b = blockIdx.y, t = threadIdx.x;
    const float* row = sim + ((long)(b * Qn + q)) * Nn;
    for (int i = t; i < Nn; i += 256) sv[i] = row[i];
    __syncthreads();
    for (int k = 0; k < Kk; ++k) {
        float bvv = -3.0e38f; int bi = 0;
        for (int i = t; i < Nn; i += 256) { float x = sv[i]; if (x > bvv) { bvv = x; bi = i; } }
        rv[t] = bvv; ri[t] = bi;
        __syncthreads();
        for (int s = 128; s > 0; s >>= 1) {
            if (t < s && rv[t + s] > rv[t]) { rv[t] = rv[t + s]; ri[t] = ri[t + s]; }
            __syncthreads();
        }
        if (t == 0) { tv[k] = rv[0]; tix[k] = ri[0]; sv[ri[0]] = -3.0e38f; }
        __syncthreads();
    }
    if (t == 0) {
        float mx = tv[0];
        for (int k = 1; k < Kk; ++k) mx = fmaxf(mx, tv[k]);
        float s = 0.f;
        for (int k = 0; k < Kk; ++k) { w[k] = expf(tv[k] - mx); s += w[k]; }
        float inv = 1.0f / s;
        for (int k = 0; k < Kk; ++k) w[k] *= inv;
    }
    __syncthreads();
    float acc = 0.f;
    for (int k = 0; k < Kk; ++k) acc += w[k] * mem[((long)b * Nn + tix[k]) * 256 + t];
    aligned[((long)(b * Qn + q)) * 256 + t] = acc;
}

// top-Kk indices of a length-Nn row (one block per batch)
__global__ __launch_bounds__(256)
void topk_idx_kernel(const float* __restrict__ imp, int* __restrict__ outIdx, int Nn, int Kk) {
    extern __shared__ unsigned char smraw2[];
    float* sv = (float*)smraw2;
    float* rv = sv + Nn;
    int*   ri = (int*)(rv + 256);
    int b = blockIdx.x, t = threadIdx.x;
    const float* row = imp + (long)b * Nn;
    for (int i = t; i < Nn; i += 256) sv[i] = row[i];
    __syncthreads();
    for (int k = 0; k < Kk; ++k) {
        float bvv = -3.0e38f; int bi = 0;
        for (int i = t; i < Nn; i += 256) { float x = sv[i]; if (x > bvv) { bvv = x; bi = i; } }
        rv[t] = bvv; ri[t] = bi;
        __syncthreads();
        for (int s = 128; s > 0; s >>= 1) {
            if (t < s && rv[t + s] > rv[t]) { rv[t] = rv[t + s]; ri[t] = ri[t + s]; }
            __syncthreads();
        }
        if (t == 0) { outIdx[(long)b * Kk + k] = ri[0]; sv[ri[0]] = -3.0e38f; }
        __syncthreads();
    }
}

__global__ void imp_max_kernel(const float* __restrict__ sim, float* __restrict__ imp,
                               int Qn, int Nn, int Bn) {
    long i = (long)blockIdx.x * 256 + threadIdx.x;
    if (i >= (long)Bn * Nn) return;
    int b = (int)(i / Nn), n = (int)(i % Nn);
    const float* base = sim + (long)b * Qn * Nn + n;
    float m = -3.0e38f;
    for (int q = 0; q < Qn; ++q) m = fmaxf(m, base[(long)q * Nn]);
    imp[i] = m;
}

__global__ __launch_bounds__(256)
void gather_rows_kernel(const float* __restrict__ mem, const int* __restrict__ idx,
                        float* __restrict__ out, int Nn, int Ck) {
    int blk = blockIdx.x;
    int b = blk / Ck, k = blk % Ck, t = threadIdx.x;
    out[(long)blk * 256 + t] = mem[((long)b * Nn + idx[(long)b * Ck + k]) * 256 + t];
}

// self attention with relation bias: one block per (q,h,b); qk rows = [qproj|kproj]
__global__ __launch_bounds__(128)
void attn_self_kernel(const float* __restrict__ qk, const float* __restrict__ v,
                      const float* __restrict__ bias, float* __restrict__ o,
                      int Qn, float scale) {
    __shared__ float sc[300];
    __shared__ float qv[32];
    __shared__ float sinv;
    int i = blockIdx.x, h = blockIdx.y, b = blockIdx.z, t = threadIdx.x;
    const float* qrow = qk + ((long)(b * Qn + i)) * 512 + h * 32;
    if (t < 32) qv[t] = qrow[t];
    __syncthreads();
    for (int j = t; j < Qn; j += 128) {
        const float* krow = qk + ((long)(b * Qn + j)) * 512 + 256 + h * 32;
        float s = 0.f;
        #pragma unroll
        for (int d = 0; d < 32; ++d) s += qv[d] * krow[d];
        sc[j] = s * scale + bias[((long)h * Qn + i) * Qn + j];
    }
    __syncthreads();
    if (t == 0) {
        float mx = -3.0e38f;
        for (int j = 0; j < Qn; ++j) mx = fmaxf(mx, sc[j]);
        float sum = 0.f;
        for (int j = 0; j < Qn; ++j) { float e = expf(sc[j] - mx); sc[j] = e; sum += e; }
        sinv = 1.0f / sum;
    }
    __syncthreads();
    if (t < 32) {
        float acc = 0.f;
        for (int j = 0; j < Qn; ++j) acc += sc[j] * v[((long)(b * Qn + j)) * 256 + h * 32 + t];
        o[((long)(b * Qn + i)) * 256 + h * 32 + t] = acc * sinv;
    }
}

// cross attention over Kn=1024 selected memory rows; kv rows = [kproj|vproj]
__global__ __launch_bounds__(256)
void attn_cross_kernel(const float* __restrict__ qc, const float* __restrict__ kv,
                       float* __restrict__ o, int Qn, int Kn, float scale) {
    __shared__ float sc[1024];
    __shared__ float qv[32];
    __shared__ float sinv;
    int i = blockIdx.x, h = blockIdx.y, b = blockIdx.z, t = threadIdx.x;
    const float* qrow = qc + ((long)(b * Qn + i)) * 256 + h * 32;
    if (t < 32) qv[t] = qrow[t];
    __syncthreads();
    for (int j = t; j < Kn; j += 256) {
        const float* krow = kv + ((long)(b * Kn + j)) * 512 + h * 32;
        float s = 0.f;
        #pragma unroll
        for (int d = 0; d < 32; ++d) s += qv[d] * krow[d];
        sc[j] = s * scale;
    }
    __syncthreads();
    if (t == 0) {
        float mx = -3.0e38f;
        for (int j = 0; j < Kn; ++j) mx = fmaxf(mx, sc[j]);
        float sum = 0.f;
        for (int j = 0; j < Kn; ++j) { float e = expf(sc[j] - mx); sc[j] = e; sum += e; }
        sinv = 1.0f / sum;
    }
    __syncthreads();
    if (t < 32) {
        float acc = 0.f;
        for (int j = 0; j < Kn; ++j) acc += sc[j] * kv[((long)(b * Kn + j)) * 512 + 256 + h * 32 + t];
        o[((long)(b * Qn + i)) * 256 + h * 32 + t] = acc * sinv;
    }
}

// q = LayerNorm(q + q2)*g + b ; one wave per row
__global__ __launch_bounds__(256)
void res_ln_kernel(float* __restrict__ q, const float* __restrict__ q2,
                   const float* __restrict__ g, const float* __restrict__ b, long rows) {
    long row = (long)blockIdx.x * 8 + (threadIdx.x >> 5);
    int lane = threadIdx.x & 31;
    if (row >= rows) return;
    float vals[8]; float s = 0.f;
    #pragma unroll
    for (int i = 0; i < 8; ++i) {
        int d = lane + 32 * i;
        float v = q[row * 256 + d] + q2[row * 256 + d];
        vals[i] = v; s += v;
    }
    float ss = 0.f;
    #pragma unroll
    for (int i = 0; i < 8; ++i) ss += vals[i] * vals[i];
    s  = wave_sum(s);
    ss = wave_sum(ss);
    float mean = s * (1.0f / 256.0f);
    float var  = ss * (1.0f / 256.0f) - mean * mean;
    float inv  = rsqrtf(var + 1e-5f);
    #pragma unroll
    for (int i = 0; i < 8; ++i) {
        int d = lane + 32 * i;
        q[row * 256 + d] = (vals[i] - mean) * inv * g[d] + b[d];
    }
}

__global__ __launch_bounds__(128)
void heads_small_kernel(const float* __restrict__ q, const float* __restrict__ clsW,
                        const float* __restrict__ clsb, const float* __restrict__ qualW,
                        const float* __restrict__ qualb, float* __restrict__ out) {
    int r = blockIdx.x * 128 + threadIdx.x;
    if (r >= 1200) return;
    const float* x = q + (long)r * 256;
    float s0 = clsb[0], s1 = clsb[1], sq = qualb[0];
    for (int d = 0; d < 256; ++d) {
        float xv = x[d];
        s0 += xv * clsW[d];
        s1 += xv * clsW[256 + d];
        sq += xv * qualW[d];
    }
    out[r * 2 + 0] = s0;
    out[r * 2 + 1] = s1;
    out[2400 + r]  = sq;
}

// ---------------------------------------------------------------------------
// Host orchestration
// ---------------------------------------------------------------------------
extern "C" void kernel_launch(void* const* d_in, const int* in_sizes, int n_in,
                              void* d_out, int out_size, void* d_ws, size_t ws_size,
                              hipStream_t stream) {
    (void)in_sizes; (void)n_in; (void)out_size; (void)ws_size;
    const int Bn = 4, Nn = 16384, Dn = 256, Qn = 300, Ln = 3;
    const int AK = 96, CK = 1024;

    const float* memory    = (const float*)d_in[0];
    const float* mask_feat = (const float*)d_in[1];
    const float* query_feat= (const float*)d_in[2];
    const float* query_pos = (const float*)d_in[3];
    const float* bato_qW = (const float*)d_in[4];
    const float* bato_mW = (const float*)d_in[5];
    const float* bato_g1W= (const float*)d_in[6];
    const float* bato_g1b= (const float*)d_in[7];
    const float* bato_g2W= (const float*)d_in[8];
    const float* bato_g2b= (const float*)d_in[9];
    const float* ref1W = (const float*)d_in[10];
    const float* ref1b = (const float*)d_in[11];
    const float* ref2W = (const float*)d_in[12];
    const float* ref2b = (const float*)d_in[13];
    const float* rel1W = (const float*)d_in[14];
    const float* rel1b = (const float*)d_in[15];
    const float* rel2W = (const float*)d_in[16];
    const float* rel2b = (const float*)d_in[17];
    const float* saWqkv= (const float*)d_in[18];
    const float* sabqkv= (const float*)d_in[19];
    const float* saWo  = (const float*)d_in[20];
    const float* sabo  = (const float*)d_in[21];
    const float* caWqkv= (const float*)d_in[22];
    const float* cabqkv= (const float*)d_in[23];
    const float* caWo  = (const float*)d_in[24];
    const float* cabo  = (const float*)d_in[25];
    const float* alqW  = (const float*)d_in[26];
    const float* almW  = (const float*)d_in[27];
    const float* ag1W  = (const float*)d_in[28];
    const float* ag1b  = (const float*)d_in[29];
    const float* ag2W  = (const float*)d_in[30];
    const float* ag2b  = (const float*)d_in[31];
    const float* cpqW  = (const float*)d_in[32];
    const float* cpmW  = (const float*)d_in[33];
    const float* f1W   = (const float*)d_in[34];
    const float* f1b   = (const float*)d_in[35];
    const float* f2W   = (const float*)d_in[36];
    const float* f2b   = (const float*)d_in[37];
    const float* n1g = (const float*)d_in[38];
    const float* n1b = (const float*)d_in[39];
    const float* n2g = (const float*)d_in[40];
    const float* n2b = (const float*)d_in[41];
    const float* n3g = (const float*)d_in[42];
    const float* n3b = (const float*)d_in[43];
    const float* clsW = (const float*)d_in[44];
    const float* clsb = (const float*)d_in[45];
    const float* qualW= (const float*)d_in[46];
    const float* qualb= (const float*)d_in[47];
    const float* efd1W= (const float*)d_in[48];
    const float* efd1b= (const float*)d_in[49];
    const float* efd2W= (const float*)d_in[50];
    const float* efd2b= (const float*)d_in[51];
    const float* me1W = (const float*)d_in[52];
    const float* me1b = (const float*)d_in[53];
    const float* me2W = (const float*)d_in[54];
    const float* me2b = (const float*)d_in[55];

    float* out = (float*)d_out;

    // ---- workspace carve (floats) ----
    float* wsp = (float*)d_ws;
    auto take = [&](size_t n) { float* p = wsp; wsp += n; return p; };
    float* mem   = take((size_t)Bn * Nn * Dn);       // 16.8M
    float* mn    = take((size_t)Bn * Nn * Dn);       // also holds BATO hidden (BN x 64)
    float* sim   = take((size_t)Bn * Qn * Nn);       // 19.7M; reused as mask_featT at the end
    float* q     = take((size_t)Bn * Qn * Dn);
    float* xb    = take((size_t)Bn * Qn * Dn);
    float* qnb   = take((size_t)Bn * Qn * Dn);
    float* alg   = take((size_t)Bn * Qn * Dn);
    float* cat   = take((size_t)Bn * Qn * 2 * Dn);
    float* qkv   = take((size_t)Bn * Qn * 512);
    float* vb    = take((size_t)Bn * Qn * Dn);
    float* ob    = take((size_t)Bn * Qn * Dn);
    float* q2b   = take((size_t)Bn * Qn * Dn);
    float* agh   = take((size_t)Bn * Qn * Dn);
    float* agb   = take((size_t)Bn * Qn * Dn);
    float* ffnh  = take((size_t)Bn * Qn * 1024);
    float* kvb   = take((size_t)Bn * CK * 512);
    float* qcb   = take((size_t)Bn * Qn * Dn);
    float* msel  = take((size_t)Bn * CK * Dn);
    float* impb  = take((size_t)Bn * Nn);
    int*   kidx  = (int*)take((size_t)Bn * CK);
    float* biasb = take((size_t)8 * Qn * Qn);
    float* rp1   = take((size_t)Qn * 128);
    float* rpb   = take(640);
    float* ub    = take(256);
    float* kerb  = take((size_t)Bn * Qn * Dn);
    float* mehb  = take((size_t)Bn * Qn * Dn);
    float* efdh  = take((size_t)Bn * Qn * 128);

    auto gemm = [&](const float* A, long bsA, long lda,
                    const float* Bm, long bsB, long ldbN,
                    float* C, long bsC, long ldc, const float* bias,
                    int M, int N, int K, int act, int batches) {
        dim3 g((unsigned)((N + 63) / 64), (unsigned)((M + 31) / 32), (unsigned)batches);
        gemm_wmma_kernel<<<g, dim3(32), 0, stream>>>(A, bsA, lda, Bm, bsB, ldbN,
                                                     C, bsC, ldc, bias, M, N, K, act);
    };

    const int BQ  = Bn * Qn;            // 1200
    const int BQD = BQ * Dn;            // 307200
    const float scale = 0.1767766952966369f; // 1/sqrt(32)

    // ---- q broadcast ----
    bcast_kernel<<<(BQD + 255) / 256, 256, 0, stream>>>(query_feat, q, BQD, Qn * Dn);

    // ---- BATO memory enhancement ----
    bato_u_kernel<<<1, 256, 0, stream>>>(query_feat, bato_qW, bato_mW, ub);
    gemm(memory, 0, Dn, bato_g1W, 0, Dn, mn, 0, 64, bato_g1b,
         Bn * Nn, 64, Dn, /*gelu*/1, 1);
    bato_apply_kernel<<<(Bn * Nn) / 8, 256, 0, stream>>>(memory, mn, bato_g2W, bato_g2b,
                                                         ub, mem, (long)Bn * Nn);

    // ---- reference points & pairwise relation bias (batch-independent) ----
    gemm(query_feat, 0, Dn, ref1W, 0, Dn, rp1, 0, 128, ref1b, Qn, 128, Dn, 1, 1);
    rp2_kernel<<<3, 256, 0, stream>>>(rp1, ref2W, ref2b, rpb);
    relbias_kernel<<<(Qn * Qn + 127) / 128, 128, 0, stream>>>(rpb, rel1W, rel1b,
                                                              rel2W, rel2b, biasb);

    size_t shA = (size_t)Nn * 4 + 256 * 4 + 256 * 4 + (size_t)AK * 12; // topk-align LDS
    size_t shI = (size_t)Nn * 4 + 256 * 4 + 256 * 4;                   // topk-idx LDS

    for (int i = 0; i < Ln; ++i) {
        const float* alqWi = alqW + (size_t)i * Dn * Dn;
        const float* almWi = almW + (size_t)i * Dn * Dn;
        const float* ag1Wi = ag1W + (size_t)i * Dn * 512;
        const float* ag1bi = ag1b + i * Dn;
        const float* ag2Wi = ag2W + (size_t)i * Dn * Dn;
        const float* ag2bi = ag2b + i * Dn;
        const float* saWqkvi = saWqkv + (size_t)i * 768 * Dn;
        const float* sabqkvi = sabqkv + i * 768;
        const float* saWoi = saWo + (size_t)i * Dn * Dn;
        const float* saboi = sabo + i * Dn;
        const float* caWqkvi = caWqkv + (size_t)i * 768 * Dn;
        const float* cabqkvi = cabqkv + i * 768;
        const float* caWoi = caWo + (size_t)i * Dn * Dn;
        const float* caboi = cabo + i * Dn;
        const float* cpqWi = cpqW + (size_t)i * Dn * Dn;
        const float* cpmWi = cpmW + (size_t)i * Dn * Dn;
        const float* f1Wi = f1W + (size_t)i * 1024 * Dn;
        const float* f1bi = f1b + i * 1024;
        const float* f2Wi = f2W + (size_t)i * Dn * 1024;
        const float* f2bi = f2b + i * Dn;

        // --- top-k query-memory alignment ---
        gemm(q, 0, Dn, alqWi, 0, Dn, qnb, 0, Dn, nullptr, BQ, Dn, Dn, 0, 1);
        l2norm_rows_kernel<<<BQ / 8, 256, 0, stream>>>(qnb, BQ);
        gemm(mem, 0, Dn, almWi, 0, Dn, mn, 0, Dn, nullptr, Bn * Nn, Dn, Dn, 0, 1);
        l2norm_rows_kernel<<<(Bn * Nn) / 8, 256, 0, stream>>>(mn, (long)Bn * Nn);
        gemm(qnb, (long)Qn * Dn, Dn, mn, (long)Nn * Dn, Dn,
             sim, (long)Qn * Nn, Nn, nullptr, Qn, Nn, Dn, 0, Bn);
        topk_align_kernel<<<dim3(Qn, Bn), 256, shA, stream>>>(sim, mem, alg, Qn, Nn, AK);
        concat_kernel<<<(BQ * 512 + 255) / 256, 256, 0, stream>>>(q, alg, cat, BQ * 512);
        gemm(cat, 0, 512, ag1Wi, 0, 512, agh, 0, Dn, ag1bi, BQ, Dn, 512, 1, 1);
        gemm(agh, 0, Dn, ag2Wi, 0, Dn, agb, 0, Dn, ag2bi, BQ, Dn, Dn, 2, 1);
        axpyg_kernel<<<(BQD + 255) / 256, 256, 0, stream>>>(q, agb, alg, BQD);

        // --- self attention with relation bias ---
        add_bcast_kernel<<<(BQD + 255) / 256, 256, 0, stream>>>(q, query_pos, xb, BQD, Qn * Dn);
        gemm(xb, 0, Dn, saWqkvi, 0, Dn, qkv, 0, 512, sabqkvi, BQ, 512, Dn, 0, 1);
        gemm(q, 0, Dn, saWqkvi + 512 * Dn, 0, Dn, vb, 0, Dn, sabqkvi + 512,
             BQ, Dn, Dn, 0, 1);
        attn_self_kernel<<<dim3(Qn, 8, Bn), 128, 0, stream>>>(qkv, vb, biasb, ob, Qn, scale);
        gemm(ob, 0, Dn, saWoi, 0, Dn, q2b, 0, Dn, saboi, BQ, Dn, Dn, 0, 1);
        res_ln_kernel<<<BQ / 8, 256, 0, stream>>>(q, q2b, n1g + i * Dn, n1b + i * Dn, BQ);

        // --- memory sparsification + cross attention ---
        gemm(q, 0, Dn, cpqWi, 0, Dn, qnb, 0, Dn, nullptr, BQ, Dn, Dn, 0, 1);
        l2norm_rows_kernel<<<BQ / 8, 256, 0, stream>>>(qnb, BQ);
        gemm(mem, 0, Dn, cpmWi, 0, Dn, mn, 0, Dn, nullptr, Bn * Nn, Dn, Dn, 0, 1);
        l2norm_rows_kernel<<<(Bn * Nn) / 8, 256, 0, stream>>>(mn, (long)Bn * Nn);
        gemm(qnb, (long)Qn * Dn, Dn, mn, (long)Nn * Dn, Dn,
             sim, (long)Qn * Nn, Nn, nullptr, Qn, Nn, Dn, 0, Bn);
        imp_max_kernel<<<(Bn * Nn) / 256, 256, 0, stream>>>(sim, impb, Qn, Nn, Bn);
        topk_idx_kernel<<<Bn, 256, shI, stream>>>(impb, kidx, Nn, CK);
        gather_rows_kernel<<<Bn * CK, 256, 0, stream>>>(mem, kidx, msel, Nn, CK);
        add_bcast_kernel<<<(BQD + 255) / 256, 256, 0, stream>>>(q, query_pos, xb, BQD, Qn * Dn);
        gemm(xb, 0, Dn, caWqkvi, 0, Dn, qcb, 0, Dn, cabqkvi, BQ, Dn, Dn, 0, 1);
        gemm(msel, 0, Dn, caWqkvi + 256 * Dn, 0, Dn, kvb, 0, 512, cabqkvi + 256,
             Bn * CK, 512, Dn, 0, 1);
        attn_cross_kernel<<<dim3(Qn, 8, Bn), 256, 0, stream>>>(qcb, kvb, ob, Qn, CK, scale);
        gemm(ob, 0, Dn, caWoi, 0, Dn, q2b, 0, Dn, caboi, BQ, Dn, Dn, 0, 1);
        res_ln_kernel<<<BQ / 8, 256, 0, stream>>>(q, q2b, n2g + i * Dn, n2b + i * Dn, BQ);

        // --- FFN ---
        gemm(q, 0, Dn, f1Wi, 0, Dn, ffnh, 0, 1024, f1bi, BQ, 1024, Dn, 1, 1);
        gemm(ffnh, 0, 1024, f2Wi, 0, 1024, q2b, 0, Dn, f2bi, BQ, Dn, 1024, 0, 1);
        res_ln_kernel<<<BQ / 8, 256, 0, stream>>>(q, q2b, n3g + i * Dn, n3b + i * Dn, BQ);
    }

    // ---- prediction heads ----
    heads_small_kernel<<<(BQ + 127) / 128, 128, 0, stream>>>(q, clsW, clsb, qualW, qualb, out);
    gemm(q, 0, Dn, efd1W, 0, Dn, efdh, 0, 128, efd1b, BQ, 128, Dn, 1, 1);
    gemm(efdh, 0, 128, efd2W, 0, 128, out + 3600, 0, 40, efd2b, BQ, 40, 128, 0, 1);
    gemm(q, 0, Dn, me1W, 0, Dn, mehb, 0, Dn, me1b, BQ, Dn, Dn, 1, 1);
    gemm(mehb, 0, Dn, me2W, 0, Dn, kerb, 0, Dn, me2b, BQ, Dn, Dn, 0, 1);

    // masks: transpose mask_feat (B,C,HW) -> (B,HW,C) into the dead sim buffer,
    // then run the same fast row-major WMMA GEMM per batch.
    float* msT = sim;  // 16.8M floats <= sim's 19.7M
    transpose_kernel<<<dim3(16384 / 32, Dn / 32, Bn), 256, 0, stream>>>(
        mask_feat, msT, Dn, 16384);
    gemm(kerb, (long)Qn * Dn, Dn,
         msT, (long)Nn * Dn, Dn,
         out + 51600, (long)Qn * 16384, 16384, nullptr,
         Qn, 16384, Dn, 0, Bn);
}
